// VisionTransformerEncoder_37555194036458
// MI455X (gfx1250) — matrix operand order repaired and checked by reading, not compiled, not run on hardware
//
#include <hip/hip_runtime.h>
#include <hip/hip_bf16.h>

// ---------------------------------------------------------------------------
// Swin transformer block (depth=2) for MI455X / gfx1250, bf16 WMMA pipeline.
// B=4, H=W=160, C=256, heads=8 (d=32), ws=10 (L=100, nW=256), hid=1024.
// ---------------------------------------------------------------------------

typedef __attribute__((ext_vector_type(16))) __bf16 v16bf;
typedef __attribute__((ext_vector_type(8)))  __bf16 v8bf;
typedef __attribute__((ext_vector_type(8)))  float  v8f;

constexpr int kB     = 4;
constexpr int kH     = 160;
constexpr int kW     = 160;
constexpr int kC     = 256;
constexpr int kHeads = 8;
constexpr int kDH    = 32;
constexpr int kWS    = 10;
constexpr int kL     = 100;   // window size^2
constexpr int kNW    = 256;   // windows per image (16x16)
constexpr int kN     = 25600; // H*W
constexpr int kHid   = 1024;
constexpr int kM     = kB * kN; // 102400 rows

__device__ __forceinline__ unsigned short f2bf(float f) {
  unsigned u = __float_as_uint(f);
  u = u + 0x7FFFu + ((u >> 16) & 1u);            // round-to-nearest-even
  return (unsigned short)(u >> 16);
}

__device__ __forceinline__ v16bf mk16(v8bf lo, v8bf hi) {
  v16bf r;
#pragma unroll
  for (int i = 0; i < 8; ++i) { r[i] = lo[i]; r[i + 8] = hi[i]; }
  return r;
}

// windowed position (w,p) with cyclic shift -> original flattened index n
__device__ __forceinline__ int win2orig(int w, int p, int shift) {
  int wr = w >> 4;            // w / 16
  int wc = w & 15;
  int pr = p / kWS, pc = p - pr * kWS;
  int r = wr * kWS + pr + shift; if (r >= kH) r -= kH;
  int c = wc * kWS + pc + shift; if (c >= kW) c -= kW;
  return r * kW + c;
}

// ---------------------------------------------------------------------------
__global__ void k_copy4(const float4* __restrict__ src, float4* __restrict__ dst, int n4) {
  int i = blockIdx.x * blockDim.x + threadIdx.x;
  if (i < n4) dst[i] = src[i];
}

// fp32 weight W[k][n] (K x Nc row-major) -> bf16 WT[n][k] (N-major for WMMA B)
__global__ void k_wt(const float* __restrict__ w, unsigned short* __restrict__ wt,
                     int K, int Nc) {
  int i = blockIdx.x * blockDim.x + threadIdx.x;
  if (i >= K * Nc) return;
  int n = i / K, k = i - n * K;
  wt[i] = f2bf(w[k * Nc + n]);
}

// ---------------------------------------------------------------------------
// LayerNorm over C=256, one wave per row, optional window-gather permutation.
__global__ void __launch_bounds__(256)
k_ln(const float* __restrict__ X, const float* __restrict__ g,
     const float* __restrict__ bt, unsigned short* __restrict__ Y,
     int shift, int win) {
  int row  = blockIdx.x * 8 + (threadIdx.x >> 5);
  int lane = threadIdx.x & 31;
  int b = row / kN, j = row - b * kN;
  int n = j;
  if (win) { int w = j / kL, p = j - w * kL; n = win2orig(w, p, shift); }
  const float* xr = X + ((size_t)b * kN + n) * kC + lane * 8;
  float4 a0 = *(const float4*)xr;
  float4 a1 = *(const float4*)(xr + 4);
  float v[8] = {a0.x, a0.y, a0.z, a0.w, a1.x, a1.y, a1.z, a1.w};
  float s = 0.f;
#pragma unroll
  for (int t = 0; t < 8; ++t) s += v[t];
#pragma unroll
  for (int m = 16; m >= 1; m >>= 1) s += __shfl_xor(s, m, 32);
  float mean = s * (1.0f / kC);
  float q = 0.f;
#pragma unroll
  for (int t = 0; t < 8; ++t) { float d = v[t] - mean; q += d * d; }
#pragma unroll
  for (int m = 16; m >= 1; m >>= 1) q += __shfl_xor(q, m, 32);
  float rstd = rsqrtf(q * (1.0f / kC) + 1e-5f);
  const float* gp = g + lane * 8;
  const float* bp = bt + lane * 8;
  union { unsigned short u[8]; uint4 q4; } o;
#pragma unroll
  for (int t = 0; t < 8; ++t)
    o.u[t] = f2bf((v[t] - mean) * rstd * gp[t] + bp[t]);
  *(uint4*)(Y + (size_t)row * kC + lane * 8) = o.q4;
}

// ---------------------------------------------------------------------------
// Tiled bf16 WMMA GEMM: out = A[M x K] @ WT^T (+ epilogue).  BM=BN=128, BK=32.
// 256 threads = 8 waves; each wave owns a 64x32 patch = 4x2 WMMA tiles.
// Double-buffered LDS: fetch k+1 into registers, compute k, commit, 1 barrier.
// mode 0: QKV  -> scatter (q*scale, k, v^T) per head          (Ncols=768)
// mode 1: proj -> X[b][inv(j)][:] += v                        (Ncols=256)
// mode 2: fc1  -> Obf = bf16(gelu(v))                         (Ncols=1024)
// mode 3: fc2  -> X[m][:] += v                                (Ncols=256)
__global__ void __launch_bounds__(256)
k_gemm(const unsigned short* __restrict__ A, const unsigned short* __restrict__ WT,
       const float* __restrict__ bias, int K, int mode, int shift,
       unsigned short* __restrict__ Q, unsigned short* __restrict__ Kg,
       unsigned short* __restrict__ VT, float* __restrict__ X,
       unsigned short* __restrict__ Obf) {
  __shared__ unsigned short As[2][128 * 40];
  __shared__ unsigned short Bs[2][128 * 40];

  const int tid  = threadIdx.x;
  const int lane = tid & 31;
  const int wave = tid >> 5;
  const int bm = blockIdx.y, bn = blockIdx.x;
  const int wm = wave >> 2, wn = wave & 3;

  v8f acc[4][2];
#pragma unroll
  for (int i = 0; i < 4; ++i)
#pragma unroll
    for (int j = 0; j < 2; ++j) acc[i][j] = (v8f){0.f,0.f,0.f,0.f,0.f,0.f,0.f,0.f};

  const int lrow  = tid >> 1;
  const int lhalf = (tid & 1) * 16;
  const size_t arow = (size_t)(bm * 128 + lrow) * K;
  const size_t brow = (size_t)(bn * 128 + lrow) * K;
  const int nk = K >> 5;

  // preload K-step 0 into buffer 0
  {
    const uint4* pa = (const uint4*)(A + arow + lhalf);
    const uint4* pb = (const uint4*)(WT + brow + lhalf);
    uint4 a0 = pa[0], a1 = pa[1], b0 = pb[0], b1 = pb[1];
    *(uint4*)(&As[0][lrow * 40 + lhalf])     = a0;
    *(uint4*)(&As[0][lrow * 40 + lhalf + 8]) = a1;
    *(uint4*)(&Bs[0][lrow * 40 + lhalf])     = b0;
    *(uint4*)(&Bs[0][lrow * 40 + lhalf + 8]) = b1;
  }
  __syncthreads();

  int cur = 0;
  for (int ks = 0; ks < nk; ++ks) {
    // fetch K-step ks+1 into registers (overlaps with WMMA below)
    uint4 na0, na1, nb0, nb1;
    const bool more = (ks + 1 < nk);
    if (more) {
      const int k1 = (ks + 1) * 32;
      const uint4* pa = (const uint4*)(A + arow + k1 + lhalf);
      const uint4* pb = (const uint4*)(WT + brow + k1 + lhalf);
      na0 = pa[0]; na1 = pa[1]; nb0 = pb[0]; nb1 = pb[1];
      if (ks + 2 < nk) {  // prefetch K-step ks+2 lines into cache
        __builtin_prefetch((const void*)(A + arow + k1 + 32 + lhalf), 0, 1);
        __builtin_prefetch((const void*)(WT + brow + k1 + 32 + lhalf), 0, 1);
      }
    }

    // compute on buffer `cur`
    const unsigned short* Ac = As[cur];
    const unsigned short* Bc = Bs[cur];
    v16bf bv[2];
#pragma unroll
    for (int ni = 0; ni < 2; ++ni) {
      int nrow = (wn * 2 + ni) * 16 + (lane & 15);
      int kb = (lane & 16) ? 8 : 0;
      bv[ni] = mk16(*(const v8bf*)(&Bc[nrow * 40 + kb]),
                    *(const v8bf*)(&Bc[nrow * 40 + kb + 16]));
    }
#pragma unroll
    for (int mi = 0; mi < 4; ++mi) {
      int mrow = (wm * 4 + mi) * 16 + (lane & 15);
      int ka = (lane & 16) ? 8 : 0;
      v16bf av = mk16(*(const v8bf*)(&Ac[mrow * 40 + ka]),
                      *(const v8bf*)(&Ac[mrow * 40 + ka + 16]));
      acc[mi][0] = __builtin_amdgcn_wmma_f32_16x16x32_bf16(
          false, av, false, bv[0], (short)0, acc[mi][0], false, false);
      acc[mi][1] = __builtin_amdgcn_wmma_f32_16x16x32_bf16(
          false, av, false, bv[1], (short)0, acc[mi][1], false, false);
    }

    // commit next tile to the other buffer
    if (more) {
      unsigned short* An = As[cur ^ 1];
      unsigned short* Bn = Bs[cur ^ 1];
      *(uint4*)(&An[lrow * 40 + lhalf])     = na0;
      *(uint4*)(&An[lrow * 40 + lhalf + 8]) = na1;
      *(uint4*)(&Bn[lrow * 40 + lhalf])     = nb0;
      *(uint4*)(&Bn[lrow * 40 + lhalf + 8]) = nb1;
    }
    __syncthreads();
    cur ^= 1;
  }

  // ---- epilogue (hoist per-column bias and per-row decomposition) ----
  const float qscale = 0.17677669529663687f; // 32^-0.5
  int   nc[2]; float bcol[2];
#pragma unroll
  for (int ni = 0; ni < 2; ++ni) {
    nc[ni]   = bn * 128 + (wn * 2 + ni) * 16 + (lane & 15);
    bcol[ni] = bias[nc[ni]];
  }
#pragma unroll
  for (int mi = 0; mi < 4; ++mi) {
#pragma unroll
    for (int r = 0; r < 8; ++r) {
      int m = bm * 128 + (wm * 4 + mi) * 16 + r + ((lane & 16) ? 8 : 0);
      int b = 0, w = 0, p = 0, norig = 0;
      if (mode == 0 || mode == 1) {
        b = m / kN; int j = m - b * kN; w = j / kL; p = j - w * kL;
        if (mode == 1) norig = win2orig(w, p, shift);
      }
#pragma unroll
      for (int ni = 0; ni < 2; ++ni) {
        float v = acc[mi][ni][r] + bcol[ni];
        int ncol = nc[ni];
        if (mode == 0) {
          int t = ncol >> 8, h = (ncol >> 5) & 7, d = ncol & 31;
          size_t base = ((size_t)(b * kNW + w) * kHeads + h) * 3200;
          if (t == 0)      Q[base + p * 32 + d]   = f2bf(v * qscale);
          else if (t == 1) Kg[base + p * 32 + d]  = f2bf(v);
          else             VT[base + d * 100 + p] = f2bf(v);
        } else if (mode == 1) {
          X[((size_t)b * kN + norig) * kC + ncol] += v;
        } else if (mode == 2) {
          float t = 0.7978845608028654f * v * (1.0f + 0.044715f * v * v);
          Obf[(size_t)m * kHid + ncol] = f2bf(0.5f * v * (1.0f + tanhf(t)));
        } else {
          X[(size_t)m * kC + ncol] += v;
        }
      }
    }
  }
}

// ---------------------------------------------------------------------------
// Fused windowed attention, one block = one (batch, window, head).
// 224 threads (7 waves): exactly one 16-row M-tile per wave in both phases.
// S = Q K^T (7x7 tiles, K=32 => 1 WMMA each), softmax with rel-pos bias +
// shift mask, O = P V (7x2 tiles, K=128 => 4 WMMA steps).
constexpr int kQsN  = 112 * 40;   // ushorts
constexpr int kVtN  = 32 * 136;
constexpr int kPsN  = 112 * 136;
constexpr int kSsN  = 112 * 116;  // floats
constexpr size_t kAttnSmem = (size_t)(kQsN * 2 + kQsN * 2 + kVtN * 2 + kPsN * 2)
                           + (size_t)kSsN * 4 + 361 * 4 + 100 * 4 + 16;

__global__ void __launch_bounds__(224)
k_attn(const unsigned short* __restrict__ Qg, const unsigned short* __restrict__ Kgl,
       const unsigned short* __restrict__ VTg, const float* __restrict__ rpb,
       unsigned short* __restrict__ Y, int shift) {
  extern __shared__ __align__(16) unsigned char smem_raw[];
  unsigned short* Qs  = (unsigned short*)smem_raw;
  unsigned short* Ks  = Qs + kQsN;
  unsigned short* VTs = Ks + kQsN;
  unsigned short* Ps  = VTs + kVtN;
  float* Ss     = (float*)(Ps + kPsN);
  float* bias_s = Ss + kSsN;
  int*   rid_s  = (int*)(bias_s + 361);

  int bid = blockIdx.x;
  int h = bid & 7;
  int w = (bid >> 3) & 255;
  int b = bid >> 11;
  int tid = threadIdx.x, lane = tid & 31, wave = tid >> 5; // wave in [0,7)

  size_t base = ((size_t)(b * kNW + w) * kHeads + h) * 3200;

  // stage Q, K (pad rows 100..111 with zeros)
  for (int t = tid; t < 112 * 4; t += 224) {
    int row = t >> 2, ch = (t & 3) * 8;
    uint4 vq = make_uint4(0, 0, 0, 0), vk = make_uint4(0, 0, 0, 0);
    if (row < kL) {
      vq = *(const uint4*)(Qg  + base + row * 32 + ch);
      vk = *(const uint4*)(Kgl + base + row * 32 + ch);
    }
    *(uint4*)(&Qs[row * 40 + ch]) = vq;
    *(uint4*)(&Ks[row * 40 + ch]) = vk;
  }
  // stage V^T [32][136], zero-pad j>=100
  for (int t = tid; t < 32 * 128; t += 224) {
    int d = t >> 7, j = t & 127;
    VTs[d * 136 + j] = (j < kL) ? VTg[base + d * 100 + j] : (unsigned short)0;
  }
  for (int t = tid; t < 361; t += 224) bias_s[t] = rpb[t * kHeads + h];
  for (int t = tid; t < kL; t += 224) {
    int rid = 0;
    if (shift > 0) {
      int wr = w >> 4, wc = w & 15;
      int pr = t / kWS, pc = t - pr * kWS;
      int r = wr * kWS + pr + shift; if (r >= kH) r -= kH;
      int c = wc * kWS + pc + shift; if (c >= kW) c -= kW;
      int rh = (r < kH - kWS) ? 0 : ((r < kH - shift) ? 1 : 2);
      int rc = (c < kW - kWS) ? 0 : ((c < kW - shift) ? 1 : 2);
      rid = rh * 3 + rc;
    }
    rid_s[t] = rid;
  }
  __syncthreads();

  // S = Q K^T : wave owns M-tile `wave`
  {
    int mt = wave;
    int mrow = mt * 16 + (lane & 15);
    int ka = (lane & 16) ? 8 : 0;
    v16bf av = mk16(*(const v8bf*)(&Qs[mrow * 40 + ka]),
                    *(const v8bf*)(&Qs[mrow * 40 + ka + 16]));
    for (int nt = 0; nt < 7; ++nt) {
      int nrow = nt * 16 + (lane & 15);
      int kb = (lane & 16) ? 8 : 0;
      v16bf bv = mk16(*(const v8bf*)(&Ks[nrow * 40 + kb]),
                      *(const v8bf*)(&Ks[nrow * 40 + kb + 16]));
      v8f c = (v8f){0.f,0.f,0.f,0.f,0.f,0.f,0.f,0.f};
      c = __builtin_amdgcn_wmma_f32_16x16x32_bf16(false, av, false, bv,
                                                  (short)0, c, false, false);
#pragma unroll
      for (int r = 0; r < 8; ++r) {
        int m = mt * 16 + r + ((lane & 16) ? 8 : 0);
        Ss[m * 116 + nrow] = c[r];
      }
    }
  }
  __syncthreads();

  // softmax per row (bias + shift mask), write P as bf16 padded to 128 cols
  if (tid < 112) {
    if (tid < kL) {
      int i = tid;
      int pri = i / kWS, pci = i - pri * kWS;
      int ridi = rid_s[i];
      float mx = -1e30f;
      for (int j = 0; j < kL; ++j) {
        int prj = j / kWS, pcj = j - prj * kWS;
        int rp = (pri - prj + 9) * 19 + (pci - pcj + 9);
        float s = Ss[i * 116 + j] + bias_s[rp];
        if (rid_s[j] != ridi) s += -1e9f;
        Ss[i * 116 + j] = s;
        mx = fmaxf(mx, s);
      }
      float sum = 0.f;
      for (int j = 0; j < kL; ++j) {
        float e = __expf(Ss[i * 116 + j] - mx);
        Ss[i * 116 + j] = e;
        sum += e;
      }
      float inv = 1.0f / sum;
      for (int j = 0; j < kL; ++j) Ps[i * 136 + j] = f2bf(Ss[i * 116 + j] * inv);
      for (int j = kL; j < 128; ++j) Ps[i * 136 + j] = 0;
    } else {
      for (int j = 0; j < 128; ++j) Ps[tid * 136 + j] = 0;
    }
  }
  __syncthreads();

  // O = P V  (K = 128 in 4 WMMA steps, N = 32 in 2 tiles); wave owns M-tile
  {
    int mt = wave;
#pragma unroll
    for (int nt = 0; nt < 2; ++nt) {
      v8f c = (v8f){0.f,0.f,0.f,0.f,0.f,0.f,0.f,0.f};
#pragma unroll
      for (int kt = 0; kt < 4; ++kt) {
        int mrow = mt * 16 + (lane & 15);
        int ka = kt * 32 + ((lane & 16) ? 8 : 0);
        v16bf av = mk16(*(const v8bf*)(&Ps[mrow * 136 + ka]),
                        *(const v8bf*)(&Ps[mrow * 136 + ka + 16]));
        int d = nt * 16 + (lane & 15);
        int kb = kt * 32 + ((lane & 16) ? 8 : 0);
        v16bf bv = mk16(*(const v8bf*)(&VTs[d * 136 + kb]),
                        *(const v8bf*)(&VTs[d * 136 + kb + 16]));
        c = __builtin_amdgcn_wmma_f32_16x16x32_bf16(false, av, false, bv,
                                                    (short)0, c, false, false);
      }
#pragma unroll
      for (int r = 0; r < 8; ++r) {
        int m = mt * 16 + r + ((lane & 16) ? 8 : 0);
        if (m < kL) {
          int col = h * kDH + nt * 16 + (lane & 15);
          Y[((size_t)b * kN + w * kL + m) * kC + col] = f2bf(c[r]);
        }
      }
    }
  }
}

// ---------------------------------------------------------------------------
extern "C" void kernel_launch(void* const* d_in, const int* in_sizes, int n_in,
                              void* d_out, int out_size, void* d_ws, size_t ws_size,
                              hipStream_t stream) {
  const float* x_in   = (const float*)d_in[0];
  const float* qkv_w  = (const float*)d_in[1];
  const float* qkv_b  = (const float*)d_in[2];
  const float* proj_w = (const float*)d_in[3];
  const float* proj_b = (const float*)d_in[4];
  const float* rpb    = (const float*)d_in[5];
  const float* ln1_g  = (const float*)d_in[6];
  const float* ln1_b  = (const float*)d_in[7];
  const float* ln2_g  = (const float*)d_in[8];
  const float* ln2_b  = (const float*)d_in[9];
  const float* fc1_w  = (const float*)d_in[10];
  const float* fc1_b  = (const float*)d_in[11];
  const float* fc2_w  = (const float*)d_in[12];
  const float* fc2_b  = (const float*)d_in[13];

  unsigned char* ws = (unsigned char*)d_ws;
  const size_t szX = (size_t)kM * kC * 4;     // 104857600
  const size_t szY = (size_t)kM * kC * 2;     // 52428800
  const size_t szH = (size_t)kM * kHid * 2;   // 209715200 (aliases QKV region)
  float*          X    = (float*)ws;
  unsigned short* Y    = (unsigned short*)(ws + szX);
  unsigned short* Qb   = (unsigned short*)(ws + szX + szY);
  unsigned short* Kb   = Qb + (size_t)kM * kC;
  unsigned short* VTb  = Kb + (size_t)kM * kC;
  unsigned short* Hbuf = Qb;                   // alias (QKV dead when used)
  unsigned short* WTbase = (unsigned short*)(ws + szX + szY + szH);
  (void)ws_size; (void)in_sizes; (void)n_in; (void)out_size;

  const size_t wtQKV = 768 * 256, wtPRJ = 256 * 256, wtFC1 = 1024 * 256, wtFC2 = 256 * 1024;
  const size_t wtLayer = wtQKV + wtPRJ + wtFC1 + wtFC2;

  for (int i = 0; i < 2; ++i) {
    unsigned short* qkvWT  = WTbase + i * wtLayer;
    unsigned short* projWT = qkvWT + wtQKV;
    unsigned short* fc1WT  = projWT + wtPRJ;
    unsigned short* fc2WT  = fc1WT + wtFC1;
    k_wt<<<(768 * 256) / 256, 256, 0, stream>>>(qkv_w + (size_t)i * 256 * 768, qkvWT, 256, 768);
    k_wt<<<(256 * 256) / 256, 256, 0, stream>>>(proj_w + (size_t)i * 256 * 256, projWT, 256, 256);
    k_wt<<<(256 * 1024) / 256, 256, 0, stream>>>(fc1_w + (size_t)i * 256 * 1024, fc1WT, 256, 1024);
    k_wt<<<(1024 * 256) / 256, 256, 0, stream>>>(fc2_w + (size_t)i * 1024 * 256, fc2WT, 1024, 256);
  }

  k_copy4<<<(kM * kC / 4) / 256, 256, 0, stream>>>((const float4*)x_in, (float4*)X, kM * kC / 4);

  for (int i = 0; i < 2; ++i) {
    int shift = (i & 1) ? (kWS / 2) : 0;
    unsigned short* qkvWT  = WTbase + i * wtLayer;
    unsigned short* projWT = qkvWT + wtQKV;
    unsigned short* fc1WT  = projWT + wtPRJ;
    unsigned short* fc2WT  = fc1WT + wtFC1;

    k_ln<<<kM / 8, 256, 0, stream>>>(X, ln1_g + i * kC, ln1_b + i * kC, Y, shift, 1);
    k_gemm<<<dim3(6, kM / 128), 256, 0, stream>>>(Y, qkvWT, qkv_b + i * 768, 256, 0, shift,
                                                  Qb, Kb, VTb, X, Y);
    k_attn<<<kB * kNW * kHeads, 224, kAttnSmem, stream>>>(Qb, Kb, VTb,
                                                          rpb + (size_t)i * 361 * kHeads, Y, shift);
    k_gemm<<<dim3(2, kM / 128), 256, 0, stream>>>(Y, projWT, proj_b + i * kC, 256, 1, shift,
                                                  Qb, Kb, VTb, X, Y);
    k_ln<<<kM / 8, 256, 0, stream>>>(X, ln2_g + i * kC, ln2_b + i * kC, Y, 0, 0);
    k_gemm<<<dim3(8, kM / 128), 256, 0, stream>>>(Y, fc1WT, fc1_b + i * kHid, 256, 2, 0,
                                                  Qb, Kb, VTb, X, Hbuf);
    k_gemm<<<dim3(2, kM / 128), 256, 0, stream>>>(Hbuf, fc2WT, fc2_b + i * kC, 1024, 3, 0,
                                                  Qb, Kb, VTb, X, Y);
  }

  k_copy4<<<(kM * kC / 4) / 256, 256, 0, stream>>>((const float4*)X, (float4*)d_out, kM * kC / 4);
}
